// GINLayer_49675591746182
// MI455X (gfx1250) — matrix operand order repaired and checked by reading, not compiled, not run on hardware
//
#include <hip/hip_runtime.h>
#include <hip/hip_bf16.h>

// ---------------------------------------------------------------------------
// GIN layer for MI455X (gfx1250, wave32, WMMA):
//   agg[dst] += x[src]   (640K edges, hardware f32 atomics, L2-resident x)
//   h   = bf16(x + agg)
//   h1  = relu(h @ W1 + b1)      -> bf16, via v_wmma_f32_16x16x32_bf16
//   out = h1 @ W2 + b2           -> f32,  via v_wmma_f32_16x16x32_bf16
// B tiles staged global->LDS with GLOBAL_LOAD_ASYNC_TO_LDS_B128 (ASYNCcnt).
// ---------------------------------------------------------------------------

typedef __attribute__((ext_vector_type(16))) __bf16 v16bf;
typedef __attribute__((ext_vector_type(8)))  float  v8f;

#define D 512
#define NNODES 20000

__device__ __forceinline__ unsigned short f2bf(float f) {
    union { float f; unsigned int u; } cvt; cvt.f = f;
    unsigned int u = cvt.u;
    unsigned int r = u + 0x7FFFu + ((u >> 16) & 1u);   // round-to-nearest-even
    return (unsigned short)(r >> 16);
}

// ---- zero the f32 aggregation buffer (graph-capture safe, no memset) ------
__global__ void gin_zero_kernel(float4* agg4, int n4) {
    int i = blockIdx.x * blockDim.x + threadIdx.x;
    if (i < n4) agg4[i] = make_float4(0.f, 0.f, 0.f, 0.f);
}

// ---- transpose + convert W1/W2 to bf16:  Wt[n][k] = bf16(W[k][n]) ---------
__global__ void gin_wcvt_kernel(const float* __restrict__ W1,
                                const float* __restrict__ W2,
                                unsigned short* __restrict__ W1t,
                                unsigned short* __restrict__ W2t) {
    int idx = blockIdx.x * blockDim.x + threadIdx.x;   // 2 * 512 * 512 threads
    int which = idx >> 18;                             // 0: W1, 1: W2
    int e = idx & ((1 << 18) - 1);
    int n = e >> 9, k = e & (D - 1);
    const float* W = which ? W2 : W1;
    unsigned short* Wt = which ? W2t : W1t;
    Wt[(size_t)n * D + k] = f2bf(W[(size_t)k * D + n]);
}

// ---- scatter-add: one thread = (edge, 4-float chunk) ----------------------
__global__ void gin_scatter_kernel(const float* __restrict__ x,
                                   const long long* __restrict__ ei,
                                   float* __restrict__ agg, int E) {
    int t = blockIdx.x * blockDim.x + threadIdx.x;
    int edge = t >> 7;            // 128 threads per edge (512 floats / 4)
    int lane = t & 127;
    if (edge >= E) return;
    int src = (int)ei[edge];
    int dst = (int)ei[(size_t)E + edge];
    const float4 v = *(const float4*)(x + (size_t)src * D + lane * 4);
    float* p = agg + (size_t)dst * D + lane * 4;
    unsafeAtomicAdd(p + 0, v.x);   // global_atomic_add_f32
    unsafeAtomicAdd(p + 1, v.y);
    unsafeAtomicAdd(p + 2, v.z);
    unsafeAtomicAdd(p + 3, v.w);
}

// ---- h = bf16(x + agg), 4 elements / thread -------------------------------
__global__ void gin_hbuild_kernel(const float* __restrict__ x,
                                  const float* __restrict__ agg,
                                  unsigned short* __restrict__ hbf, int n4) {
    int i = blockIdx.x * blockDim.x + threadIdx.x;
    if (i >= n4) return;
    const float4 xv = ((const float4*)x)[i];
    const float4 av = ((const float4*)agg)[i];
    unsigned int lo = (unsigned int)f2bf(xv.x + av.x) |
                      ((unsigned int)f2bf(xv.y + av.y) << 16);
    unsigned int hi = (unsigned int)f2bf(xv.z + av.z) |
                      ((unsigned int)f2bf(xv.w + av.w) << 16);
    ((uint2*)hbf)[i] = make_uint2(lo, hi);
}

// ---- bf16 WMMA GEMM: out[M,512] = act(A[M,512] @ Bt^T + bias) -------------
// A row-major bf16, Bt is B transposed ([n][k], bf16). One wave -> 16x64 tile.
// Block = 256 threads = 8 waves = 8 consecutive M-tiles; blockIdx.y = 64-col tile.
// B tile (64 n x 32 k, 4KB) staged via async global->LDS, one B128 per thread.
template <bool RELU_BF16_OUT>
__global__ __launch_bounds__(256) void gin_gemm_kernel(
        const unsigned short* __restrict__ A,
        const unsigned short* __restrict__ Bt,
        const float* __restrict__ bias,
        void* __restrict__ out, int M) {
    __shared__ unsigned short sB[64 * 32];            // 64 cols x 32 k, 4KB
    unsigned int* sB32 = (unsigned int*)sB;

    const int tid  = threadIdx.x;
    const int wave = tid >> 5;
    const int lane = tid & 31;
    const int half = lane >> 4;                       // 0: lanes 0-15, 1: 16-31
    const int lr   = lane & 15;
    const int mtile = blockIdx.x * 8 + wave;
    const bool active = (mtile * 16) < M;             // wave-uniform
    const int m0 = mtile * 16;
    const int n0 = blockIdx.y * 64;

    union Frag { unsigned int u[8]; v16bf v; };

    v8f c[4];
    #pragma unroll
    for (int sub = 0; sub < 4; ++sub) {
        float bv = bias[n0 + sub * 16 + lr];          // C(m,n) = b[n], all m
        #pragma unroll
        for (int g = 0; g < 8; ++g) c[sub][g] = bv;
    }

    // per-thread async-staging addresses: thread stages dwords [tid*4, tid*4+3]
    const int srow = tid >> 2;                        // 0..63  (n within tile)
    const int scw  = (tid & 3) * 4;                   // dword within 16-dword row
    const unsigned int ldsDst =
        (unsigned int)(size_t)(sB32 + srow * 16 + scw);   // LDS byte address

    const unsigned short* arow = A + (size_t)(m0 + lr) * D;

    for (int kk = 0; kk < D; kk += 32) {
        // async global -> LDS: 256 threads x 16B = 4KB tile (ASYNCcnt)
        const unsigned short* gsrc = Bt + (size_t)(n0 + srow) * D + kk + scw * 2;
        asm volatile("global_load_async_to_lds_b128 %0, %1, off"
                     :: "v"(ldsDst), "v"(gsrc) : "memory");
        asm volatile("s_wait_asynccnt 0" ::: "memory");
        __syncthreads();

        if (active) {
            // prefetch next K-slice of this A row into caches
            if (kk + 32 < D)
                __builtin_prefetch(arow + kk + 32, 0, 3);

            // A fragment: 16x32 bf16, lane holds row m0+lr;
            // VGPR g (pair p): k = 2g + (g>=4 ? 8 : 0) + half*8
            Frag a;
            #pragma unroll
            for (int g = 0; g < 8; ++g) {
                int k = kk + 2 * g + ((g >= 4) ? 8 : 0) + half * 8;
                a.u[g] = *(const unsigned int*)(arow + k);
            }
            // preload all 4 B fragments (distinct regs) so the 4 WMMAs can
            // issue back-to-back with operand-A reuse
            Frag b[4];
            #pragma unroll
            for (int sub = 0; sub < 4; ++sub) {
                int nl = sub * 16 + lr;
                #pragma unroll
                for (int g = 0; g < 8; ++g)
                    b[sub].u[g] = sB32[nl * 16 + g + ((g >= 4) ? 4 : 0) + half * 4];
            }
            #pragma unroll
            for (int sub = 0; sub < 4; ++sub)
                c[sub] = __builtin_amdgcn_wmma_f32_16x16x32_bf16(
                    false, a.v, false, b[sub].v, (short)0, c[sub], false, false);
        }
        __syncthreads();
    }

    if (active) {
        // D layout: VGPR g -> m = g + half*8, n = n0 + sub*16 + lr
        #pragma unroll
        for (int sub = 0; sub < 4; ++sub) {
            int n = n0 + sub * 16 + lr;
            #pragma unroll
            for (int g = 0; g < 8; ++g) {
                int m = m0 + g + half * 8;
                float v = c[sub][g];
                if (RELU_BF16_OUT) {
                    v = v > 0.f ? v : 0.f;
                    ((unsigned short*)out)[(size_t)m * D + n] = f2bf(v);
                } else {
                    ((float*)out)[(size_t)m * D + n] = v;
                }
            }
        }
    }
}

// ---------------------------------------------------------------------------
extern "C" void kernel_launch(void* const* d_in, const int* in_sizes, int n_in,
                              void* d_out, int out_size, void* d_ws, size_t ws_size,
                              hipStream_t stream) {
    const float*      x  = (const float*)d_in[0];
    const long long*  ei = (const long long*)d_in[1];   // int64 [2, E]
    const float*      W1 = (const float*)d_in[2];
    const float*      b1 = (const float*)d_in[3];
    const float*      W2 = (const float*)d_in[4];
    const float*      b2 = (const float*)d_in[5];
    float* out = (float*)d_out;
    const int E = in_sizes[1] / 2;                      // 640000

    // workspace layout (bytes)
    char* ws = (char*)d_ws;
    float*          agg  = (float*)ws;                               // 40,960,000
    unsigned short* hbf  = (unsigned short*)(ws + 40960000);         // 20,480,000
    unsigned short* h1bf = (unsigned short*)(ws + 61440000);         // 20,480,000
    unsigned short* W1t  = (unsigned short*)(ws + 81920000);         //    524,288
    unsigned short* W2t  = (unsigned short*)(ws + 82444288);         //    524,288

    const int nelem  = NNODES * D;        // 10,240,000
    const int n4     = nelem / 4;         //  2,560,000

    gin_zero_kernel<<<(n4 + 255) / 256, 256, 0, stream>>>((float4*)agg, n4);
    gin_wcvt_kernel<<<(2 * D * D) / 256, 256, 0, stream>>>(W1, W2, W1t, W2t);
    gin_scatter_kernel<<<(E * 128 + 255) / 256, 256, 0, stream>>>(x, ei, agg, E);
    gin_hbuild_kernel<<<(n4 + 255) / 256, 256, 0, stream>>>(x, agg, hbf, n4);

    dim3 ggrid((NNODES / 16 + 7) / 8, D / 64);          // (157, 8)
    gin_gemm_kernel<true ><<<ggrid, 256, 0, stream>>>(hbf,  W1t, b1, h1bf, NNODES);
    gin_gemm_kernel<false><<<ggrid, 256, 0, stream>>>(h1bf, W2t, b2, out,  NNODES);
}
